// CaspNetSubsampling_37950331028043
// MI455X (gfx1250) — compile-verified, matrix operand order checked by reading, not run on hardware
//
#include <hip/hip_runtime.h>
#include <hip/hip_bf16.h>

typedef __attribute__((ext_vector_type(16))) _Float16 v16h;
typedef __attribute__((ext_vector_type(8)))  _Float16 v8h;
typedef __attribute__((ext_vector_type(8)))  float    v8f;
typedef __attribute__((ext_vector_type(4)))  float    v4f;

// ---- problem sizes (fixed by reference) ----
#define BATCH 16
#define XH 100
#define XW 64
#define C1H 92
#define C1W 56
#define NC  256
#define PH  42
#define PW  24
#define K1P 96          // conv1 K = 81 padded to 3*32
#define K2  20736       // pc conv K = 9*9*256
#define M1  82432       // BATCH*C1H*C1W
#define M2  16128       // BATCH*PH*PW
#define ICAPS 32256     // PH*PW*32
#define NJ 10
#define NWCH 512        // routing i-chunks (one wave each)
#define IPW 63          // ICAPS / NWCH

// pcconv blocking
#define NROW_B 64       // B panel rows (n) staged in LDS per block
#define LDS_STR 264     // halves per LDS row: 256 + 8 pad (TDM pad: 16B per 512B)

#if __has_builtin(__builtin_amdgcn_tensor_load_to_lds)
#define HAVE_TDM 1
#else
#define HAVE_TDM 0
#endif

// =====================================================================
// weight conversion: conv1_w [81][256] f32 -> w1t [256][96] f16 (padded)
// =====================================================================
__global__ void k_cvt_w1(const float* __restrict__ w1, _Float16* __restrict__ w1t) {
    int tid = blockIdx.x * 256 + threadIdx.x;      // 256*96 = 24576 threads
    int n = tid / K1P, K = tid - (tid / K1P) * K1P;
    float v = (K < 81) ? w1[K * NC + n] : 0.0f;
    w1t[n * K1P + K] = (_Float16)v;
}

// pc_w [20736][256] f32 -> w2t [256][20736] f16 (coalesced writes, L2-hit reads)
__global__ void k_cvt_w2(const float* __restrict__ w2, _Float16* __restrict__ w2t) {
    long tid = (long)blockIdx.x * 256 + threadIdx.x;   // 256*20736 threads
    int n = (int)(tid / K2);
    int k = (int)(tid - (long)n * K2);
    w2t[tid] = (_Float16)w2[(size_t)k * NC + n];
}

__global__ void k_zero(float* __restrict__ p, int nelem) {
    int t = blockIdx.x * 256 + threadIdx.x;
    if (t < nelem) p[t] = 0.0f;
}

// =====================================================================
// conv1: 9x9, 1->256, stride 1, valid, +bias, relu. WMMA f16 implicit GEMM.
// M = 82432 (b,oh,ow), N = 256, K = 81 (pad 96). One wave per 16x16 tile.
// Branchless A build: tap index clamped; padded-K garbage multiplies the
// zero-padded B columns, so the product is exactly 0.
// =====================================================================
__global__ void __launch_bounds__(256)
k_conv1(const float* __restrict__ x, const _Float16* __restrict__ w1t,
        const float* __restrict__ b1, _Float16* __restrict__ y) {
    int wid  = (blockIdx.x * blockDim.x + threadIdx.x) >> 5;
    int lane = threadIdx.x & 31;
    int mtile = wid >> 4;           // 5152 m-tiles
    int ntile = wid & 15;           // 16 n-tiles
    int half  = lane >> 4;
    int lmod  = lane & 15;

    int mrow = mtile * 16 + lmod;
    int b    = mrow / (C1H * C1W);
    int rem  = mrow - b * (C1H * C1W);
    int oh   = rem / C1W;
    int ow   = rem - oh * C1W;
    const float* xb = x + (size_t)b * XH * XW;

    int n = ntile * 16 + lmod;
    const _Float16* brow = w1t + (size_t)n * K1P;

    v8f acc = {};
#pragma unroll
    for (int kk = 0; kk < K1P; kk += 32) {
        v16h a;
#pragma unroll
        for (int e = 0; e < 8; ++e) {
            int K0 = kk + half * 8 + e;            // elements 0..7
            int K1 = kk + 16 + half * 8 + e;       // elements 8..15
            int K0c = (K0 < 81) ? K0 : 80;         // clamp: B is zero there
            int K1c = (K1 < 81) ? K1 : 80;
            int kh0 = K0c / 9, kw0 = K0c - kh0 * 9;
            int kh1 = K1c / 9, kw1 = K1c - kh1 * 9;
            a[e]     = (_Float16)xb[(oh + kh0) * XW + (ow + kw0)];
            a[e + 8] = (_Float16)xb[(oh + kh1) * XW + (ow + kw1)];
        }
        v8h blo = *(const v8h*)(brow + kk + half * 8);
        v8h bhi = *(const v8h*)(brow + kk + 16 + half * 8);
        v16h bf = __builtin_shufflevector(blo, bhi, 0,1,2,3,4,5,6,7,8,9,10,11,12,13,14,15);
        acc = __builtin_amdgcn_wmma_f32_16x16x32_f16(false, a, false, bf, (short)0, acc, false, false);
    }

    float bias = b1[n];
#pragma unroll
    for (int v = 0; v < 8; ++v) {
        int mr = mtile * 16 + v + half * 8;
        float val = fmaxf(acc[v] + bias, 0.0f);
        y[(size_t)mr * NC + n] = (_Float16)val;
    }
}

// =====================================================================
// TDM: stage one B weight panel [NROW_B rows x 256 halves] into LDS.
// 2D tensor: dim0 = K (contiguous, 20736 halves), dim1 = n rows,
// dim0 stride = 20736. LDS padding: 16B after every 512B row (bank spread).
// =====================================================================
#if HAVE_TDM
__device__ __forceinline__ void tdm_load_b_panel(unsigned int lds_addr, const void* gaddr) {
    typedef unsigned int u32x4 __attribute__((ext_vector_type(4)));
    typedef int i32x4 __attribute__((ext_vector_type(4)));
    typedef int i32x8 __attribute__((ext_vector_type(8)));
    unsigned long long ga = (unsigned long long)(size_t)gaddr;
    u32x4 g0 = { 1u,                                   // count=1 (valid D#)
                 lds_addr,                             // lds_addr
                 (unsigned int)ga,                     // global_addr[31:0]
                 (unsigned int)((ga >> 32) & 0x1FFFFFFu) | (2u << 30) }; // addr[56:32] | type=2
    i32x8 g1 = {
        (int)((1u << 16) | (1u << 20) | (6u << 22) | (3u << 25)),
        //     data_size=2B  pad_en     every 512B    pad 16B
        (int)(((unsigned)K2 & 0xFFFFu) << 16),          // tensor_dim0[15:0]
        (int)((((unsigned)K2 >> 16) & 0xFFFFu) | (256u << 16)), // dim0[31:16] | tensor_dim1[15:0]=256
        (int)(256u << 16),                              // dim1[31:16]=0 | tile_dim0=256
        (int)((unsigned)NROW_B),                        // tile_dim1=64, tile_dim2=0
        (int)K2,                                        // tensor_dim0_stride[31:0]
        0, 0                                            // stride hi / dim1_stride (unused, 2D)
    };
    i32x4 gz = {0, 0, 0, 0};
#if __clang_major__ >= 23
    i32x8 gz8 = {0, 0, 0, 0, 0, 0, 0, 0};
    __builtin_amdgcn_tensor_load_to_lds(g0, g1, gz, gz, gz8, 0);
#else
    __builtin_amdgcn_tensor_load_to_lds(g0, g1, gz, gz, 0);
#endif
    __builtin_amdgcn_s_wait_tensorcnt(0);
}
#endif

// =====================================================================
// PrimaryCaps conv: 9x9, 256->256, stride 2, valid, +bias. 171 GFLOP.
// Implicit GEMM M=16128, N=256, K=20736. Block = 128M x 64N, 8 waves,
// each wave 2x2 WMMA tiles. B panel TDM-staged in LDS per tap.
// =====================================================================
__global__ void __launch_bounds__(256)
k_pcconv(const _Float16* __restrict__ y, const _Float16* __restrict__ w2t,
         const float* __restrict__ b2, float* __restrict__ p) {
    __shared__ _Float16 lbs[NROW_B * LDS_STR];     // 33.8 KB

    int mblk = blockIdx.x >> 2;        // 126 m-blocks of 128 rows
    int nblk = blockIdx.x & 3;         // 4 n-panels of 64
    int w    = threadIdx.x >> 5;
    int lane = threadIdx.x & 31;
    int wm   = w >> 1;                 // 0..3 -> 32-row slices
    int wn   = w & 1;                  // 0..1 -> 32-col slices
    int half = lane >> 4;
    int lmod = lane & 15;

    // per-lane A row decode for this wave's two m-tiles
    const _Float16* ybase[2];
    int ohw[2];
#pragma unroll
    for (int mt = 0; mt < 2; ++mt) {
        int m   = mblk * 128 + wm * 32 + mt * 16 + lmod;
        int b   = m / (PH * PW);
        int rem = m - b * (PH * PW);
        int oh  = rem / PW;
        int ow  = rem - oh * PW;
        ybase[mt] = y + (size_t)b * C1H * C1W * NC;
        ohw[mt]   = (2 * oh) * C1W + 2 * ow;
    }
    int nrow[2];
#pragma unroll
    for (int nt = 0; nt < 2; ++nt) nrow[nt] = wn * 32 + nt * 16 + lmod;

    const _Float16* gpanel = w2t + (size_t)(nblk * NROW_B) * K2;
    unsigned int ldsoff = (unsigned int)(size_t)(&lbs[0]);

    v8f acc[2][2] = {};

    for (int t = 0; t < 81; ++t) {
        __syncthreads();                            // previous panel consumed
#if HAVE_TDM
        if (threadIdx.x < 32)                       // wave 0 drives the DMA
            tdm_load_b_panel(ldsoff, (const void*)(gpanel + t * NC));
#else
        for (int idx = threadIdx.x; idx < NROW_B * 32; idx += 256) {
            int row = idx >> 5, c8 = (idx & 31) * 8;
            *(v8h*)&lbs[row * LDS_STR + c8] = *(const v8h*)(gpanel + (size_t)row * K2 + t * NC + c8);
        }
#endif
        __syncthreads();                            // panel ready

        int kh = t / 9, kw = t - (t / 9) * 9;
#pragma unroll
        for (int cc = 0; cc < NC; cc += 32) {
            v16h a[2], bf[2];
#pragma unroll
            for (int mt = 0; mt < 2; ++mt) {
                const _Float16* ar = ybase[mt] + (size_t)(ohw[mt] + kh * C1W + kw) * NC;
                v8h alo = *(const v8h*)(ar + cc + half * 8);
                v8h ahi = *(const v8h*)(ar + cc + 16 + half * 8);
                a[mt] = __builtin_shufflevector(alo, ahi, 0,1,2,3,4,5,6,7,8,9,10,11,12,13,14,15);
            }
#pragma unroll
            for (int nt = 0; nt < 2; ++nt) {
                const _Float16* br = &lbs[nrow[nt] * LDS_STR];
                v8h blo = *(const v8h*)(br + cc + half * 8);
                v8h bhi = *(const v8h*)(br + cc + 16 + half * 8);
                bf[nt] = __builtin_shufflevector(blo, bhi, 0,1,2,3,4,5,6,7,8,9,10,11,12,13,14,15);
            }
#pragma unroll
            for (int mt = 0; mt < 2; ++mt)
#pragma unroll
                for (int nt = 0; nt < 2; ++nt)
                    acc[mt][nt] = __builtin_amdgcn_wmma_f32_16x16x32_f16(
                        false, a[mt], false, bf[nt], (short)0, acc[mt][nt], false, false);
        }
    }

#pragma unroll
    for (int mt = 0; mt < 2; ++mt)
#pragma unroll
        for (int nt = 0; nt < 2; ++nt) {
            int n = nblk * NROW_B + wn * 32 + nt * 16 + lmod;
            float bias = b2[n];
#pragma unroll
            for (int v = 0; v < 8; ++v) {
                int mr = mblk * 128 + wm * 32 + mt * 16 + v + half * 8;
                p[(size_t)mr * NC + n] = acc[mt][nt][v] + bias;
            }
        }
}

// =====================================================================
// squash capsules in place: 516096 groups of 8 contiguous floats
// =====================================================================
__global__ void k_squash(float* __restrict__ p) {
    long g = (long)blockIdx.x * 256 + threadIdx.x;
    float* q = p + g * 8;
    v4f a = *(const v4f*)q;
    v4f c = *(const v4f*)(q + 4);
    float sq = a.x*a.x + a.y*a.y + a.z*a.z + a.w*a.w
             + c.x*c.x + c.y*c.y + c.z*c.z + c.w*c.w;
    float scale = sq / (1.0f + sq) / sqrtf(sq + 1e-7f);
    a *= scale; c *= scale;
    *(v4f*)q = a;
    *(v4f*)(q + 4) = c;
}

// =====================================================================
// routing iteration: recompute u_hat from caps_W streaming (L2-resident),
// logits from V_acc (b_r = (sum of prior v)·u_hat), softmax over J,
// accumulate s partials. One wave per 63-capsule chunk; lanes own the
// 160 (j,d) pairs (5 each) -> 5-register accumulators per batch.
// =====================================================================
__global__ void __launch_bounds__(256)
k_routing(const float* __restrict__ u, const float* __restrict__ Wc,
          const float* __restrict__ Vacc, float* __restrict__ spart, int first) {
    __shared__ float vsh[BATCH * NJ * 16];   // 2560 floats: V_acc staged in LDS
    for (int t = threadIdx.x; t < BATCH * NJ * 16; t += 256) vsh[t] = Vacc[t];
    __syncthreads();

    int wid  = (blockIdx.x * 256 + threadIdx.x) >> 5;   // 0..511
    int lane = threadIdx.x & 31;
    int i0   = wid * IPW;

    float acc[BATCH][5];
#pragma unroll
    for (int b = 0; b < BATCH; ++b)
#pragma unroll
        for (int q = 0; q < 5; ++q) acc[b][q] = 0.0f;

    for (int ii = 0; ii < IPW; ++ii) {
        int i = i0 + ii;
        float wq[5][8];
#pragma unroll
        for (int q = 0; q < 5; ++q) {
            int pp = q * 32 + lane;
            const float* wp = Wc + ((size_t)(pp >> 4) * ICAPS + i) * 128 + (size_t)(pp & 15) * 8;
            v4f w0 = *(const v4f*)wp;
            v4f w1 = *(const v4f*)(wp + 4);
            wq[q][0] = w0.x; wq[q][1] = w0.y; wq[q][2] = w0.z; wq[q][3] = w0.w;
            wq[q][4] = w1.x; wq[q][5] = w1.y; wq[q][6] = w1.z; wq[q][7] = w1.w;
            if (ii + 1 < IPW)
                __builtin_prefetch((const void*)(wp + 128), 0, 0);
        }
#pragma unroll
        for (int b = 0; b < BATCH; ++b) {
            const float* up = u + ((size_t)b * ICAPS + i) * 8;
            v4f u0 = *(const v4f*)up;
            v4f u1 = *(const v4f*)(up + 4);
            float uk[8] = {u0.x, u0.y, u0.z, u0.w, u1.x, u1.y, u1.z, u1.w};

            float uh[5];
#pragma unroll
            for (int q = 0; q < 5; ++q) {
                float s = 0.0f;
#pragma unroll
                for (int k = 0; k < 8; ++k) s = fmaf(wq[q][k], uk[k], s);
                uh[q] = s;
            }

            float c[5];
            if (first) {
#pragma unroll
                for (int q = 0; q < 5; ++q) c[q] = 0.1f;   // softmax of zeros
            } else {
                float e[5]; float z = 0.0f;
#pragma unroll
                for (int q = 0; q < 5; ++q) {
                    int pp = q * 32 + lane;
                    float pl = vsh[b * 160 + pp] * uh[q];
#pragma unroll
                    for (int off = 1; off < 16; off <<= 1) pl += __shfl_xor(pl, off, 32);
                    e[q] = __expf(pl);
                    z += e[q];
                }
#pragma unroll
                for (int off = 1; off < 32; off <<= 1) z += __shfl_xor(z, off, 32);
                float inv = 16.0f / z;
#pragma unroll
                for (int q = 0; q < 5; ++q) c[q] = e[q] * inv;
            }
#pragma unroll
            for (int q = 0; q < 5; ++q) acc[b][q] = fmaf(c[q], uh[q], acc[b][q]);
        }
    }

    float* sp = spart + (size_t)wid * (BATCH * 160);
#pragma unroll
    for (int b = 0; b < BATCH; ++b)
#pragma unroll
        for (int q = 0; q < 5; ++q)
            sp[b * 160 + q * 32 + lane] = acc[b][q];
}

// =====================================================================
// reduce partials (fixed order -> deterministic), v = squash(s),
// V_acc += v; last iteration writes d_out
// =====================================================================
__global__ void k_update(const float* __restrict__ spart, float* __restrict__ Vacc,
                         float* __restrict__ out, int last) {
    int t = blockIdx.x * 256 + threadIdx.x;       // 0..2559 over (b, j*16+d)
    float s = 0.0f;
    for (int w = 0; w < NWCH; ++w) s += spart[(size_t)w * (BATCH * 160) + t];

    __shared__ float sm[256];
    sm[threadIdx.x] = s * s;
    __syncthreads();
    int base = (threadIdx.x >> 4) << 4;
    float sq = 0.0f;
#pragma unroll
    for (int d = 0; d < 16; ++d) sq += sm[base + d];

    float scale = sq / (1.0f + sq) / sqrtf(sq + 1e-7f);
    float v = scale * s;
    Vacc[t] += v;
    if (last) out[t] = v;
}

// =====================================================================
extern "C" void kernel_launch(void* const* d_in, const int* in_sizes, int n_in,
                              void* d_out, int out_size, void* d_ws, size_t ws_size,
                              hipStream_t stream) {
    const float* x  = (const float*)d_in[0];
    const float* w1 = (const float*)d_in[1];
    const float* b1 = (const float*)d_in[2];
    const float* w2 = (const float*)d_in[3];
    const float* b2 = (const float*)d_in[4];
    const float* Wc = (const float*)d_in[5];
    float* out = (float*)d_out;

    char* ws = (char*)d_ws;
    auto carve = [&](size_t bytes) -> char* {
        char* r = ws;
        ws += (bytes + 255) & ~(size_t)255;
        return r;
    };
    _Float16* w1t  = (_Float16*)carve((size_t)256 * K1P * 2);          //  48 KB
    _Float16* w2t  = (_Float16*)carve((size_t)256 * K2 * 2);           //  10.6 MB
    _Float16* y    = (_Float16*)carve((size_t)M1 * NC * 2);            //  42.2 MB
    float*    p    = (float*)carve((size_t)M2 * NC * 4);               //  16.5 MB (-> u in place)
    float*    spart= (float*)carve((size_t)NWCH * BATCH * 160 * 4);    //   5.2 MB
    float*    vacc = (float*)carve((size_t)BATCH * NJ * 16 * 4);       //  10 KB

    k_cvt_w1<<<96, 256, 0, stream>>>(w1, w1t);
    k_cvt_w2<<<K2, 256, 0, stream>>>(w2, w2t);
    k_zero  <<<10, 256, 0, stream>>>(vacc, BATCH * NJ * 16);

    k_conv1 <<<10304, 256, 0, stream>>>(x, w1t, b1, y);   // 82432 waves / 8 per block
    k_pcconv<<<504,   256, 0, stream>>>(y, w2t, b2, p);   // 126 m-blocks x 4 n-panels
    k_squash<<<2016,  256, 0, stream>>>(p);               // 516096 capsule groups

    for (int r = 0; r < 3; ++r) {
        k_routing<<<64, 256, 0, stream>>>(p, Wc, vacc, spart, (r == 0) ? 1 : 0);
        k_update <<<10, 256, 0, stream>>>(spart, vacc, out, (r == 2) ? 1 : 0);
    }
}
